// HeteGAT_multi_RL4_56633438765562
// MI455X (gfx1250) — compile-verified, matrix operand order checked by reading, not run on hardware
//
#include <hip/hip_runtime.h>

typedef __attribute__((ext_vector_type(16))) __bf16 v16bf;
typedef __attribute__((ext_vector_type(8)))  __bf16 v8bf;
typedef __attribute__((ext_vector_type(8)))  float  v8f;

#define R_   3
#define H_   256
#define N0_  50000
#define N1_  10000
#define N2_  4000
#define N3_  1024
#define FAN_ 16

// ---------------------------------------------------------------------------
// Transpose + convert weights: W (K x N, f32, row-major) -> Wt (N x K, bf16)
// so that WMMA B-fragments become two contiguous 16B loads per lane.
// ---------------------------------------------------------------------------
__global__ void transpose_w_kernel(const float* __restrict__ W,
                                   __bf16* __restrict__ Wt, int K, int N)
{
    int idx = blockIdx.x * blockDim.x + threadIdx.x;
    if (idx >= K * N) return;
    int k = idx / N, n = idx - k * N;
    Wt[(size_t)n * K + k] = (__bf16)W[idx];
}

// Raw A data for one K-step: 2 runs of 8 f32 (ISA 16-bit fragment layout).
struct ARaw { float4 f0, f1, f2, f3; };

static __device__ __forceinline__ ARaw load_araw(const float* __restrict__ p)
{
    ARaw r;
    const float4* a0 = reinterpret_cast<const float4*>(p);
    const float4* a1 = reinterpret_cast<const float4*>(p + 16);
    r.f0 = a0[0]; r.f1 = a0[1];
    r.f2 = a1[0]; r.f3 = a1[1];
    return r;
}

static __device__ __forceinline__ v16bf cvt_afrag(const ARaw& r)
{
    v16bf a;
    a[0]  = (__bf16)r.f0.x; a[1]  = (__bf16)r.f0.y;
    a[2]  = (__bf16)r.f0.z; a[3]  = (__bf16)r.f0.w;
    a[4]  = (__bf16)r.f1.x; a[5]  = (__bf16)r.f1.y;
    a[6]  = (__bf16)r.f1.z; a[7]  = (__bf16)r.f1.w;
    a[8]  = (__bf16)r.f2.x; a[9]  = (__bf16)r.f2.y;
    a[10] = (__bf16)r.f2.z; a[11] = (__bf16)r.f2.w;
    a[12] = (__bf16)r.f3.x; a[13] = (__bf16)r.f3.y;
    a[14] = (__bf16)r.f3.z; a[15] = (__bf16)r.f3.w;
    return a;
}

// B fragment: two contiguous 16B bf16 loads.
static __device__ __forceinline__ v16bf load_bfrag(const __bf16* __restrict__ p)
{
    const v8bf* q0 = reinterpret_cast<const v8bf*>(p);
    const v8bf* q1 = reinterpret_cast<const v8bf*>(p + 16);
    v8bf g0 = q0[0], g1 = q1[0];
    v16bf b;
#pragma unroll
    for (int i = 0; i < 8; ++i) { b[i] = g0[i]; b[8 + i] = g1[i]; }
    return b;
}

// ---------------------------------------------------------------------------
// WMMA GEMM: Y[M x N] = X[rowmap][M x K](f32) * Wt(N x K, bf16)^T + bias
// Block = 4 waves; wave w owns rows [64*by + 16*w, +16) and a 16x64 column
// strip (4 accumulators): one A fragment feeds FOUR v_wmma_f32_16x16x32_bf16
// per K-step.  Explicit register double-buffering: next K-step's A-raw + 4 B
// fragments are issued BEFORE the current step's WMMAs, so the s_wait in
// front of the matrix ops only covers loads issued a full iteration earlier.
// act: 0 = none, 1 = tanh epilogue.  Requires N % 64 == 0, M % 16 == 0.
// ---------------------------------------------------------------------------
__launch_bounds__(128)
__global__ void gemm_wmma_kernel(const float* __restrict__ X,
                                 const int* __restrict__ rowidx,
                                 const __bf16* __restrict__ Wt,
                                 const float* __restrict__ bias,
                                 float* __restrict__ Y,
                                 int M, int N, int K, int act)
{
    const int lane   = threadIdx.x & 31;
    const int wave   = threadIdx.x >> 5;
    const int m_base = blockIdx.y * 64 + wave * 16;
    if (m_base >= M) return;                 // whole-wave exit only
    const int n_base = blockIdx.x * 64;
    const int half   = lane >> 4;
    const int l15    = lane & 15;
    int mrow = m_base + l15;
    if (mrow >= M) mrow = M - 1;             // clamp (M is x16 anyway)
    const int arow = rowidx ? rowidx[mrow] : mrow;
    const float*  __restrict__ ap = X  + (size_t)arow * K + 8 * half;
    const __bf16* __restrict__ bp = Wt + (size_t)(n_base + l15) * K + 8 * half;

    // ---- prologue: stage K-step 0 ----
    ARaw  ar = load_araw(ap);
    v16bf b0 = load_bfrag(bp);
    v16bf b1 = load_bfrag(bp + (size_t)16 * K);
    v16bf b2 = load_bfrag(bp + (size_t)32 * K);
    v16bf b3 = load_bfrag(bp + (size_t)48 * K);

    v8f acc0 = {}, acc1 = {}, acc2 = {}, acc3 = {};
    for (int kk = 0; kk < K; kk += 32) {
        const int kn = kk + 32;
        // ---- issue next K-step's loads first (stay in flight past WMMAs) ----
        ARaw  arn = ar;
        v16bf b0n = b0, b1n = b1, b2n = b2, b3n = b3;
        if (kn < K) {
            arn = load_araw(ap + kn);
            const __bf16* bpk = bp + kn;
            b0n = load_bfrag(bpk);
            b1n = load_bfrag(bpk + (size_t)16 * K);
            b2n = load_bfrag(bpk + (size_t)32 * K);
            b3n = load_bfrag(bpk + (size_t)48 * K);
        }

        // ---- consume current K-step ----
        v16bf afrag = cvt_afrag(ar);
        acc0 = __builtin_amdgcn_wmma_f32_16x16x32_bf16(false, afrag, false, b0,
                                                       (short)0, acc0, false, false);
        acc1 = __builtin_amdgcn_wmma_f32_16x16x32_bf16(false, afrag, false, b1,
                                                       (short)0, acc1, false, false);
        acc2 = __builtin_amdgcn_wmma_f32_16x16x32_bf16(false, afrag, false, b2,
                                                       (short)0, acc2, false, false);
        acc3 = __builtin_amdgcn_wmma_f32_16x16x32_bf16(false, afrag, false, b3,
                                                       (short)0, acc3, false, false);

        ar = arn; b0 = b0n; b1 = b1n; b2 = b2n; b3 = b3n;
    }

    v8f accs[4] = {acc0, acc1, acc2, acc3};
#pragma unroll
    for (int t = 0; t < 4; ++t) {
        const int ncol = n_base + 16 * t + l15;
        const float bn = bias ? bias[ncol] : 0.0f;
#pragma unroll
        for (int vv = 0; vv < 8; ++vv) {
            int m = m_base + vv + 8 * half;
            if (m < M) {
                float val = accs[t][vv] + bn;
                if (act == 1) val = tanhf(val);
                Y[(size_t)m * N + ncol] = val;
            }
        }
    }
}

// ---------------------------------------------------------------------------
// Neighbor attention: dst node j owns edges [j*16, (j+1)*16). One wave per
// (dst, head): 16 dot products, register softmax, weighted V sum.
// ---------------------------------------------------------------------------
__launch_bounds__(32)
__global__ void attn_agg_kernel(const float* __restrict__ q,
                                const float* __restrict__ k,
                                const float* __restrict__ v,
                                const int* __restrict__ src,
                                float* __restrict__ out, int heads, int ch)
{
    const int dst  = blockIdx.x;
    const int h    = blockIdx.y;
    const int lane = threadIdx.x;
    const int hc   = heads * ch;
    const float scale = rsqrtf((float)ch);
    const float* __restrict__ qp = q + (size_t)dst * hc + h * ch;
    const int*   __restrict__ ep = src + (size_t)dst * FAN_;

    int   srcs[FAN_];
    float logit[FAN_];
#pragma unroll
    for (int e = 0; e < FAN_; ++e) {
        const int s = ep[e];
        srcs[e] = s;
        const float* kp = k + (size_t)s * hc + h * ch;
        float p = 0.f;
        for (int c = lane; c < ch; c += 32) p += qp[c] * kp[c];
#pragma unroll
        for (int off = 16; off > 0; off >>= 1) p += __shfl_xor(p, off, 32);
        logit[e] = p * scale;
    }
    float m = logit[0];
#pragma unroll
    for (int e = 1; e < FAN_; ++e) m = fmaxf(m, logit[e]);
    float a[FAN_]; float z = 0.f;
#pragma unroll
    for (int e = 0; e < FAN_; ++e) { a[e] = __expf(logit[e] - m); z += a[e]; }
    const float inv = 1.f / z;
    for (int c = lane; c < ch; c += 32) {
        float accv = 0.f;
#pragma unroll
        for (int e = 0; e < FAN_; ++e)
            accv += a[e] * v[(size_t)srcs[e] * hc + h * ch + c];
        out[(size_t)dst * hc + h * ch + c] = accv * inv;
    }
}

// ---------------------------------------------------------------------------
// Gated skip: g = sigmoid([out, xr, out-xr] . Wb);  y = g*xr + (1-g)*out
// ---------------------------------------------------------------------------
__launch_bounds__(128)
__global__ void gate_kernel(const float* __restrict__ attn,
                            const float* __restrict__ xr,
                            const float* __restrict__ Wb,
                            float* __restrict__ y, int hc)
{
    const int dst = blockIdx.x;
    const int tid = threadIdx.x;
    __shared__ float red[4];
    const float* op = attn + (size_t)dst * hc;
    const float* xp = xr   + (size_t)dst * hc;
    float p = 0.f;
    for (int j = tid; j < hc; j += 128) {
        float o = op[j], x = xp[j];
        p += o * Wb[j] + x * Wb[hc + j] + (o - x) * Wb[2 * hc + j];
    }
#pragma unroll
    for (int off = 16; off > 0; off >>= 1) p += __shfl_xor(p, off, 32);
    if ((tid & 31) == 0) red[tid >> 5] = p;
    __syncthreads();
    const float tot = red[0] + red[1] + red[2] + red[3];
    const float g = 1.f / (1.f + __expf(-tot));
    for (int j = tid; j < hc; j += 128) {
        float o = op[j], x = xp[j];
        y[(size_t)dst * hc + j] = g * x + (1.f - g) * o;
    }
}

// ---------------------------------------------------------------------------
// Per-column (feature) sums for mean/var across nodes, then normalize + ELU.
// ---------------------------------------------------------------------------
__global__ void colsum_kernel(const float* __restrict__ x,
                              float* __restrict__ sums, float* __restrict__ sqs,
                              int nrows, int rows_per_block)
{
    const int c  = threadIdx.x;                    // 256 columns
    const int r0 = blockIdx.x * rows_per_block;
    int r1 = r0 + rows_per_block; if (r1 > nrows) r1 = nrows;
    float s = 0.f, q = 0.f;
    for (int r = r0; r < r1; ++r) { float val = x[(size_t)r * H_ + c]; s += val; q += val * val; }
    atomicAdd(&sums[c], s);
    atomicAdd(&sqs[c], q);
}

__global__ void norm_elu_kernel(float* __restrict__ x,
                                const float* __restrict__ sums,
                                const float* __restrict__ sqs, int nrows)
{
    const int total = nrows * H_;
    for (int idx = blockIdx.x * blockDim.x + threadIdx.x; idx < total;
         idx += gridDim.x * blockDim.x) {
        const int c = idx & (H_ - 1);
        const float mean = sums[c] / nrows;
        const float var  = sqs[c] / nrows - mean * mean;
        float vv = (x[idx] - mean) * rsqrtf(var + 1e-5f);
        x[idx] = vv > 0.f ? vv : __expf(vv) - 1.f;
    }
}

// ---------------------------------------------------------------------------
// multi[node][r][c] = emb[r][node][c] * rl[r]
// ---------------------------------------------------------------------------
__global__ void build_multi_kernel(const float* __restrict__ emb,
                                   const float* __restrict__ rl,
                                   float* __restrict__ multi)
{
    const int total = N3_ * R_ * H_;
    for (int idx = blockIdx.x * blockDim.x + threadIdx.x; idx < total;
         idx += gridDim.x * blockDim.x) {
        const int c = idx & (H_ - 1);
        const int r = (idx >> 8) % R_;
        const int node = idx / (R_ * H_);
        multi[idx] = emb[((size_t)r * N3_ + node) * H_ + c] * rl[r];
    }
}

// ---------------------------------------------------------------------------
// Final relation attention: s_r = v[node,r,:] . u ; softmax over r; mix multi.
// blockDim = 256 = H_, one block per node.
// ---------------------------------------------------------------------------
__launch_bounds__(256)
__global__ void final_attn_kernel(const float* __restrict__ multi,
                                  const float* __restrict__ vt,
                                  const float* __restrict__ u,
                                  float* __restrict__ out)
{
    const int node = blockIdx.x;
    const int tid  = threadIdx.x;
    __shared__ float red[R_][8];
    const float uval = u[tid];
#pragma unroll
    for (int r = 0; r < R_; ++r) {
        float p = vt[((size_t)node * R_ + r) * H_ + tid] * uval;
#pragma unroll
        for (int off = 16; off > 0; off >>= 1) p += __shfl_xor(p, off, 32);
        if ((tid & 31) == 0) red[r][tid >> 5] = p;
    }
    __syncthreads();
    float s[R_];
#pragma unroll
    for (int r = 0; r < R_; ++r) {
        float t = 0.f;
#pragma unroll
        for (int w = 0; w < 8; ++w) t += red[r][w];
        s[r] = t;
    }
    float m  = fmaxf(s[0], fmaxf(s[1], s[2]));
    float e0 = __expf(s[0] - m), e1 = __expf(s[1] - m), e2 = __expf(s[2] - m);
    float inv = 1.f / (e0 + e1 + e2);
    float acc = multi[((size_t)node * R_ + 0) * H_ + tid] * e0
              + multi[((size_t)node * R_ + 1) * H_ + tid] * e1
              + multi[((size_t)node * R_ + 2) * H_ + tid] * e2;
    out[(size_t)node * H_ + tid] = acc * inv;
}

// ---------------------------------------------------------------------------
// Host-side orchestration
// ---------------------------------------------------------------------------
extern "C" void kernel_launch(void* const* d_in, const int* in_sizes, int n_in,
                              void* d_out, int out_size, void* d_ws, size_t ws_size,
                              hipStream_t stream)
{
    (void)in_sizes; (void)n_in; (void)out_size; (void)ws_size;

    const float* features = (const float*)d_in[0];
    const int*   n_ids    = (const int*)d_in[1];
    const int*   srcE[3]  = {(const int*)d_in[2], (const int*)d_in[4], (const int*)d_in[6]};

    // setup_inputs dict order per layer: Wq, Wk, Wv, Ws, bq, bk, bv, bs, Wb
    const float* W[3][4]; const float* B[3][4]; const float* Wb[3];
    int base = 8;
    for (int l = 0; l < 3; ++l) {
        for (int i = 0; i < 4; ++i) W[l][i] = (const float*)d_in[base + i];
        for (int i = 0; i < 4; ++i) B[l][i] = (const float*)d_in[base + 4 + i];
        Wb[l] = (const float*)d_in[base + 8];
        base += 9;
    }
    const float* w_omega = (const float*)d_in[35];
    const float* b_omega = (const float*)d_in[36];
    const float* u_omega = (const float*)d_in[37];
    const float* rl      = (const float*)d_in[38];

    char* ws = (char*)d_ws;
    size_t off = 0;
    auto carve = [&](size_t bytes) -> void* {
        void* p = ws + off;
        off += (bytes + 255) & ~(size_t)255;
        return p;
    };
    __bf16* wtbuf[4];
    for (int i = 0; i < 4; ++i) wtbuf[i] = (__bf16*)carve((size_t)256 * 256 * sizeof(__bf16));
    __bf16* wtomega = (__bf16*)carve((size_t)256 * 256 * sizeof(__bf16));
    float* qbuf    = (float*)carve((size_t)N1_ * H_ * 4);
    float* kbuf    = (float*)carve((size_t)N0_ * H_ * 4);
    float* vbuf    = (float*)carve((size_t)N0_ * H_ * 4);
    float* xrbuf   = (float*)carve((size_t)N1_ * H_ * 4);
    float* attnbuf = (float*)carve((size_t)N1_ * H_ * 4);
    float* x1      = (float*)carve((size_t)N1_ * H_ * 4);
    float* x2      = (float*)carve((size_t)N2_ * 128 * 4);
    float* emb     = (float*)carve((size_t)R_ * N3_ * H_ * 4);
    float* multi   = (float*)carve((size_t)N3_ * R_ * H_ * 4);
    float* vtanh   = (float*)carve((size_t)N3_ * R_ * H_ * 4);
    float* colsum  = (float*)carve((size_t)H_ * 4);
    float* colsq   = (float*)carve((size_t)H_ * 4);

    auto prep = [&](const float* Wsrc, __bf16* Wt, int K, int N) {
        int tot = K * N;
        hipLaunchKernelGGL(transpose_w_kernel, dim3((tot + 255) / 256), dim3(256), 0, stream,
                           Wsrc, Wt, K, N);
    };
    auto gemm = [&](const float* X, const int* rowidx, const __bf16* Wt, const float* bias,
                    float* Y, int M, int N, int K, int act) {
        dim3 g(N / 64, (M + 63) / 64);
        hipLaunchKernelGGL(gemm_wmma_kernel, g, dim3(128), 0, stream,
                           X, rowidx, Wt, bias, Y, M, N, K, act);
    };

    const int din[3]   = {H_, H_, 128};
    const int hcv[3]   = {H_, 128, H_};
    const int heads[3] = {2, 1, 1};
    const int ndst[3]  = {N1_, N2_, N3_};
    const int nsrc[3]  = {N0_, N1_, N2_};

    for (int r = 0; r < R_; ++r) {
        const int* nid_r = n_ids + (size_t)r * N0_;
        for (int l = 0; l < 3; ++l) {
            const int K = din[l], hc = hcv[l];
            for (int i = 0; i < 4; ++i)
                prep(W[l][i] + (size_t)r * K * hc, wtbuf[i], K, hc);
            const int*   ridx = (l == 0) ? nid_r : nullptr;
            const float* Xl   = (l == 0) ? features : ((l == 1) ? x1 : x2);

            gemm(Xl, ridx, wtbuf[0], B[l][0] + (size_t)r * hc, qbuf,  ndst[l], hc, K, 0);
            gemm(Xl, ridx, wtbuf[1], B[l][1] + (size_t)r * hc, kbuf,  nsrc[l], hc, K, 0);
            gemm(Xl, ridx, wtbuf[2], B[l][2] + (size_t)r * hc, vbuf,  nsrc[l], hc, K, 0);
            gemm(Xl, ridx, wtbuf[3], B[l][3] + (size_t)r * hc, xrbuf, ndst[l], hc, K, 0);

            const int* src_r = srcE[l] + (size_t)r * ndst[l] * FAN_;
            hipLaunchKernelGGL(attn_agg_kernel, dim3(ndst[l], heads[l]), dim3(32), 0, stream,
                               qbuf, kbuf, vbuf, src_r, attnbuf, heads[l], hc / heads[l]);

            float* yout = (l == 0) ? x1 : ((l == 1) ? x2 : (emb + (size_t)r * N3_ * H_));
            hipLaunchKernelGGL(gate_kernel, dim3(ndst[l]), dim3(128), 0, stream,
                               attnbuf, xrbuf, Wb[l] + (size_t)r * 3 * hc, yout, hc);

            if (l == 0) {
                hipMemsetAsync(colsum, 0, H_ * 4, stream);
                hipMemsetAsync(colsq,  0, H_ * 4, stream);
                hipLaunchKernelGGL(colsum_kernel, dim3((N1_ + 63) / 64), dim3(H_), 0, stream,
                                   x1, colsum, colsq, N1_, 64);
                hipLaunchKernelGGL(norm_elu_kernel, dim3(512), dim3(256), 0, stream,
                                   x1, colsum, colsq, N1_);
            }
        }
    }

    hipLaunchKernelGGL(build_multi_kernel, dim3(256), dim3(256), 0, stream, emb, rl, multi);
    prep(w_omega, wtomega, H_, H_);
    gemm(multi, nullptr, wtomega, b_omega, vtanh, N3_ * R_, H_, H_, 1);
    hipLaunchKernelGGL(final_attn_kernel, dim3(N3_), dim3(H_), 0, stream,
                       multi, vtanh, u_omega, (float*)d_out);
}
